// RNNModel_62259845923161
// MI455X (gfx1250) — matrix operand order repaired and checked by reading, not compiled, not run on hardware
//
#include <hip/hip_runtime.h>
#include <stdint.h>

// Problem constants from the reference
#define RNN_B 4096
#define RNN_T 512
#define RNN_H 32

typedef __attribute__((ext_vector_type(16))) __bf16 v16bf;
typedef __attribute__((ext_vector_type(8)))  float  v8f;

static __device__ __forceinline__ float act_tanh(float x) {
#if __has_builtin(__builtin_amdgcn_tanhf)
  return __builtin_amdgcn_tanhf(x);   // v_tanh_f32 on gfx1250
#else
  return tanhf(x);
#endif
}

// Pack two f32 -> one dword {hi:bf16, lo:bf16} in exactly one instruction.
// (The compiler already emits this mnemonic, so the assembler accepts it.)
static __device__ __forceinline__ uint32_t pack2_bf16(float lo, float hi) {
  uint32_t r;
  asm("v_cvt_pk_bf16_f32 %0, %1, %2" : "=v"(r) : "v"(lo), "v"(hi));
  return r;
}

// Cross-row exchange for the D->B repack:
//   out_a = { a.lanes[0:15],  b.lanes[0:15]  }   (B slot p)
//   out_b = { a.lanes[16:31], b.lanes[16:31] }   (B slot p+4)
// v_permlane16_swap_b32 does this in ONE VALU op; fallback uses
// ds_swizzle(SWAPX16) + cndmask.
static __device__ __forceinline__ void row_exchange(bool low, uint32_t t0,
                                                    uint32_t t1,
                                                    uint32_t& out_lo,
                                                    uint32_t& out_hi) {
#if __has_builtin(__builtin_amdgcn_permlane16_swap)
  typedef __attribute__((ext_vector_type(2))) unsigned int v2u;
  v2u r = __builtin_amdgcn_permlane16_swap(t0, t1, false, false);
  out_lo = r.x;   // new t0 = {t0.row0, t1.row0}
  out_hi = r.y;   // new t1 = {t0.row1, t1.row1}
#else
  const uint32_t mine = low ? t1 : t0;  // value my partner lane needs
  const uint32_t sw =
      (uint32_t)__builtin_amdgcn_ds_swizzle((int)mine, 0x401F);  // xor 16
  out_lo = low ? t0 : sw;
  out_hi = low ? sw : t1;
#endif
}

static __device__ __forceinline__ float f4_elem(const float4& v, int s) {
  return (s == 0) ? v.x : (s == 1) ? v.y : (s == 2) ? v.z : v.w;
}

union BfPack { v16bf v; uint32_t u[8]; };

// One wave owns TWO independent 16-row batch tiles and runs the full T-step
// recurrence for both, interleaved:
//   h_new^T = tanh( W_hh x h^T  +  (x_t * W_ih + b) )
// A (16x32 bf16)  = W_hh rows (two tiles: j=0..15, j=16..31), constant/shared.
// B (32x16 bf16)  = h^T (lane n = batch col). D (16x16 f32) = h_new^T.
// D-layout -> next B-layout needs only a row0/row1 exchange per packed pair.
// Two chains per wave fill the WMMA->VALU hazard slots and the tanh/pack
// windows with the other chain's work (each wave likely runs alone on a SIMD).
__global__ __launch_bounds__(128) void rnn_wmma_kernel(
    const float* __restrict__ x,     // [B,T]
    const float* __restrict__ W_ih,  // [H,1]
    const float* __restrict__ W_hh,  // [H,H]
    const float* __restrict__ b_ih,  // [H]
    const float* __restrict__ b_hh,  // [H]
    const float* __restrict__ fc_w,  // [1,H]
    const float* __restrict__ fc_b,  // [1]
    float* __restrict__ out)         // [B]
{
  const int lane  = threadIdx.x & 31;
  const int wave  = threadIdx.x >> 5;
  const int wtile = blockIdx.x * 4 + wave;  // 128 wave-tiles of 32 batch rows
  const int lo    = lane & 15;
  const bool low  = lane < 16;

  // ---- A operands: W_hh as two 16x32 bf16 tiles (shared by both chains) ----
  // A layout used here (16x32, contiguous K, same order as B): lanes 0-15 hold M=lane, K in {0..15};
  // lanes 16-31 hold M=lane-16, K in {16..31}; pairs per VGPR.
  BfPack a0, a1;
  {
    const int m0 = lo;
    const int m1 = 16 + lo;
#pragma unroll
    for (int v = 0; v < 8; ++v) {
      const int k0 = (low ? 0 : 16) + 2 * v;
      a0.u[v] = pack2_bf16(W_hh[m0 * RNN_H + k0], W_hh[m0 * RNN_H + k0 + 1]);
      a1.u[v] = pack2_bf16(W_hh[m1 * RNN_H + k0], W_hh[m1 * RNN_H + k0 + 1]);
    }
  }

  // ---- per-lane input-projection constants (C/D layout: lane=batch col) ----
  // C/D 16x16 f32: lane<16 -> rows M=0..7, lane>=16 -> rows M=8..15.
  const int hb = low ? 0 : 8;
  float wih0[8], wih1[8], bia0[8], bia1[8];
#pragma unroll
  for (int r = 0; r < 8; ++r) {
    wih0[r] = W_ih[hb + r];
    wih1[r] = W_ih[16 + hb + r];
    bia0[r] = b_ih[hb + r]      + b_hh[hb + r];
    bia1[r] = b_ih[16 + hb + r] + b_hh[16 + hb + r];
  }

  // ---- recurrence state for the two chains ---------------------------------
  const float* xrow[2];
  xrow[0] = x + (size_t)(wtile * 32 + lo) * RNN_T;
  xrow[1] = xrow[0] + (size_t)16 * RNN_T;

  BfPack bmat[2];
  float h0v[2][8], h1v[2][8];
#pragma unroll
  for (int c = 0; c < 2; ++c) {
#pragma unroll
    for (int i = 0; i < 8; ++i) bmat[c].u[i] = 0u;
#pragma unroll
    for (int r = 0; r < 8; ++r) { h0v[c][r] = 0.f; h1v[c][r] = 0.f; }
  }

  // Software pipeline: x for iteration tq+1 is loaded before computing tq,
  // so the global-load latency overlaps 8 recurrence chain-steps of compute.
  float4 cur[2], nxt[2];
  cur[0] = *(const float4*)(xrow[0]);
  cur[1] = *(const float4*)(xrow[1]);

#pragma unroll 1
  for (int tq = 0; tq < RNN_T / 4; ++tq) {
    const int tn = (tq + 1 < RNN_T / 4) ? (tq + 1) : tq;
    nxt[0] = *(const float4*)(xrow[0] + 4 * tn);
    nxt[1] = *(const float4*)(xrow[1] + 4 * tn);
    __builtin_prefetch(xrow[0] + 4 * tq + 64, 0, 0);  // global_prefetch_b8
    __builtin_prefetch(xrow[1] + 4 * tq + 64, 0, 0);

#pragma unroll
    for (int s = 0; s < 4; ++s) {
#pragma unroll
      for (int c = 0; c < 2; ++c) {
        const float xv = f4_elem(cur[c], s);

        // C operand = x_t * W_ih + (b_ih + b_hh), already in D layout
        v8f c0, c1;
#pragma unroll
        for (int r = 0; r < 8; ++r) {
          c0[r] = fmaf(xv, wih0[r], bia0[r]);
          c1[r] = fmaf(xv, wih1[r], bia1[r]);
        }

        // D = W_hh_tile x h^T + C  (two 16x16 output tiles cover j=0..31)
        v8f d0 = __builtin_amdgcn_wmma_f32_16x16x32_bf16(
            false, a0.v, false, bmat[c].v, (short)0, c0, false, false);
        v8f d1 = __builtin_amdgcn_wmma_f32_16x16x32_bf16(
            false, a1.v, false, bmat[c].v, (short)0, c1, false, false);

#pragma unroll
        for (int r = 0; r < 8; ++r) {
          h0v[c][r] = act_tanh(d0[r]);
          h1v[c][r] = act_tanh(d1[r]);
        }

        // Repack D -> next-step B operand: pack tanh results to bf16 pairs,
        // then one row-exchange per pair puts everything in B layout.
        //   slot p   : K = 2p   (low lanes, own t0) / 16+2p (high, partner t1)
        //   slot p+4 : K = 8+2p (low, partner t0)   / 24+2p (high, own t1)
#pragma unroll
        for (int p = 0; p < 4; ++p) {
          const uint32_t t0 = pack2_bf16(h0v[c][2 * p], h0v[c][2 * p + 1]);
          const uint32_t t1 = pack2_bf16(h1v[c][2 * p], h1v[c][2 * p + 1]);
          row_exchange(low, t0, t1, bmat[c].u[p], bmat[c].u[p + 4]);
        }
      }
    }
    cur[0] = nxt[0];
    cur[1] = nxt[1];
  }

  // ---- final projection: out[b] = h_last . fc_w + fc_b ---------------------
#pragma unroll
  for (int c = 0; c < 2; ++c) {
    float acc = 0.f;
#pragma unroll
    for (int r = 0; r < 8; ++r) {
      acc = fmaf(h0v[c][r], fc_w[hb + r], acc);
      acc = fmaf(h1v[c][r], fc_w[16 + hb + r], acc);
    }
    // lane L holds half the hidden dims for batch (base + L%16); fold partner.
    const int swi = __builtin_amdgcn_ds_swizzle(__float_as_int(acc), 0x401F);
    acc += __int_as_float(swi);
    if (low) out[wtile * 32 + c * 16 + lo] = acc + fc_b[0];
  }
}

extern "C" void kernel_launch(void* const* d_in, const int* in_sizes, int n_in,
                              void* d_out, int out_size, void* d_ws, size_t ws_size,
                              hipStream_t stream) {
  (void)in_sizes; (void)n_in; (void)out_size; (void)d_ws; (void)ws_size;
  const float* x    = (const float*)d_in[0];
  const float* W_ih = (const float*)d_in[1];
  const float* W_hh = (const float*)d_in[2];
  const float* b_ih = (const float*)d_in[3];
  const float* b_hh = (const float*)d_in[4];
  const float* fc_w = (const float*)d_in[5];
  const float* fc_b = (const float*)d_in[6];
  float* out = (float*)d_out;

  // 128 waves total: 32 batch rows (2 chains) per wave, 4 waves per block.
  dim3 grid(RNN_B / (32 * 4));
  dim3 block(128);
  rnn_wmma_kernel<<<grid, block, 0, stream>>>(x, W_ih, W_hh, b_ih, b_hh,
                                              fc_w, fc_b, out);
}
